// WorldModel_77472620085748
// MI455X (gfx1250) — compile-verified, matrix operand order checked by reading, not run on hardware
//
#include <hip/hip_runtime.h>
#include <math.h>

// ---------------------------------------------------------------------------
// DreamerV3-style world model forward for MI455X (gfx1250), wave32 + WMMA.
// Bandwidth-bound on the 802MB recon_logits write; all matmuls use
// v_wmma_f32_16x16x32_f16 with f16-packed weights/activations.
// ---------------------------------------------------------------------------

#define DEV __device__ __forceinline__
#define CEILDIV(a, b) (((a) + (b) - 1) / (b))

typedef _Float16 h16;
typedef __attribute__((ext_vector_type(16))) _Float16 v16h;
typedef __attribute__((ext_vector_type(8)))  _Float16 v8h;
typedef __attribute__((ext_vector_type(8)))  float    v8f;

// Model dims
#define TT 8
#define BB 8
#define CC 3
#define HH 64
#define WW 64
#define AA 18
#define LL 32
#define KK 32
#define DD 512
#define EE 1024
#define FEAT 1536
#define BINS 255
#define NN 64   // T*B

// ---------------------------------------------------------------------------
// WMMA fragment load: 16-bit A/B 16x32 layout.
// lane holds row (M or N) = lane&15; K element e maps to
//   k = 16*(e>>3) + 8*(lane>>4) + (e&7)
// => per lane: two contiguous 8-half (16B) chunks at +8*(lane>>4) and +16.
// ---------------------------------------------------------------------------
DEV v16h load_frag(const h16* __restrict__ rowk, int lane) {
  const h16* p = rowk + ((lane >> 4) << 3);
  v8h lo = *(const v8h*)(p);
  v8h hi = *(const v8h*)(p + 16);
  v16h r;
#pragma unroll
  for (int i = 0; i < 8; ++i) { r[i] = lo[i]; r[8 + i] = hi[i]; }
  return r;
}

// ---------------------------------------------------------------------------
// Generic GEMM: Out[M,N] = act(A[M,Kp](f16) * W[N,Kp](f16)^T + bias[n>>bshift])
// One wave computes one 16x16 tile. A rows padded to ceil16(M), W rows to
// ceil16(N), K padded to mult of 32 with zeros (done by pack kernels).
// act: 0=none, 1=relu, 2=sigmoid
// ---------------------------------------------------------------------------
__global__ __launch_bounds__(256) void gemm_wmma(
    const h16* __restrict__ A, int lda,
    const h16* __restrict__ W, int ldw,
    const float* __restrict__ bias, int bshift,
    float* __restrict__ Out, int ldo,
    int M, int N, int Kp, int act)
{
  const int lane = threadIdx.x & 31;
  const int wave = blockIdx.x * (blockDim.x >> 5) + (threadIdx.x >> 5);
  const int tilesN = (N + 15) >> 4;
  const int tilesM = (M + 15) >> 4;
  if (wave >= tilesM * tilesN) return;   // wave-uniform: EXEC stays all-1s
  const int tm = wave / tilesN;
  const int tn = wave - tm * tilesN;

  const h16* arow = A + (size_t)(tm * 16 + (lane & 15)) * lda;
  const h16* wrow = W + (size_t)(tn * 16 + (lane & 15)) * ldw;

  v8f acc = {};
  for (int k = 0; k < Kp; k += 32) {
    v16h af = load_frag(arow + k, lane);
    v16h bf = load_frag(wrow + k, lane);
    acc = __builtin_amdgcn_wmma_f32_16x16x32_f16(
        false, af, false, bf, (short)0, acc, false, false);
  }

  const int n = tn * 16 + (lane & 15);
  if (n >= N) return;
  const float bv = bias ? bias[n >> bshift] : 0.f;
  const int mbase = tm * 16 + ((lane >> 4) << 3);
#pragma unroll
  for (int v = 0; v < 8; ++v) {
    int m = mbase + v;
    if (m >= M) continue;
    float x = acc[v] + bv;
    if (act == 1)      x = fmaxf(x, 0.f);
    else if (act == 2) x = 1.f / (1.f + expf(-x));
    Out[(size_t)m * ldo + n] = x;
  }
}

// ---------------------------------------------------------------------------
// Final 1x1 conv (765 out-ch) fused with [N,C,H,W,255] permute.
// A4[262144,32] f16 (rows = (img,y,x)), Wc[768,32] f16, bias[765].
// Each wave blocks 4 N-tiles so its A fragment is loaded once and reused by
// 4 WMMAs (A L2 traffic /4; the HBM store stream is the roofline).
// dst = ((img*3 + c)*4096 + pix)*255 + bin, c=n/255, bin=n%255.
// ---------------------------------------------------------------------------
#define RNT 4   // N-tiles per wave (48 tiles / 4 = 12 groups)
__global__ __launch_bounds__(256) void gemm_recon(
    const h16* __restrict__ A, const h16* __restrict__ W,
    const float* __restrict__ bias, float* __restrict__ Out)
{
  const int lane = threadIdx.x & 31;
  const int wave = blockIdx.x * (blockDim.x >> 5) + (threadIdx.x >> 5);
  const int groupsN = 48 / RNT;             // 12
  const int tilesM = 262144 / 16;           // 16384
  if (wave >= tilesM * groupsN) return;
  const int tm = wave / groupsN;
  const int gn = wave - tm * groupsN;

  const h16* arow = A + (size_t)(tm * 16 + (lane & 15)) * 32;
  v16h af = load_frag(arow, lane);

  v8f acc[RNT];
#pragma unroll
  for (int j = 0; j < RNT; ++j) {
    const int tn = gn * RNT + j;
    const h16* wrow = W + (size_t)(tn * 16 + (lane & 15)) * 32;
    v16h bf = load_frag(wrow, lane);
    v8f c = {};
    acc[j] = __builtin_amdgcn_wmma_f32_16x16x32_f16(
        false, af, false, bf, (short)0, c, false, false);
  }

  const int mbase = tm * 16 + ((lane >> 4) << 3);
#pragma unroll
  for (int j = 0; j < RNT; ++j) {
    const int n = (gn * RNT + j) * 16 + (lane & 15);
    if (n >= CC * BINS) continue;
    const int c = n / BINS, bin = n - c * BINS;
    const float bv = bias[n];
#pragma unroll
    for (int v = 0; v < 8; ++v) {
      int m = mbase + v;
      int img = m >> 12, pix = m & 4095;
      size_t dst = ((size_t)(img * 3 + c) * 4096 + (size_t)pix) * 255 + bin;
      Out[dst] = acc[j][v] + bv;
    }
  }
}

// ---------------------------------------------------------------------------
// Weight pack: fp32 src (element (n,k) at n*sn + k*sk) -> f16 dst[Np,Kp],
// zero padded. Handles row-major FC weights and deconv (ic,oc,ky,kx) weights.
// ---------------------------------------------------------------------------
__global__ void pack_w(const float* __restrict__ src, h16* __restrict__ dst,
                       int Nn, int Kd, int Np, int Kp, int sn, int sk)
{
  int idx = blockIdx.x * blockDim.x + threadIdx.x;
  if (idx >= Np * Kp) return;
  int np = idx / Kp, kp = idx - np * Kp;
  float v = (np < Nn && kp < Kd) ? src[(size_t)np * sn + (size_t)kp * sk] : 0.f;
  dst[idx] = (h16)v;
}

// fp32 [Ms,Ns] (row stride lds) -> f16 [Mp,ldh], zero padded
__global__ void cvt_pad(const float* __restrict__ src, int Ms, int Ns, int lds,
                        h16* __restrict__ dst, int Mp, int ldh)
{
  int idx = blockIdx.x * blockDim.x + threadIdx.x;
  if (idx >= Mp * ldh) return;
  int m = idx / ldh, c = idx - m * ldh;
  float v = (m < Ms && c < Ns) ? src[(size_t)m * lds + c] : 0.f;
  dst[idx] = (h16)v;
}

// ---------------------------------------------------------------------------
// Direct k=4,s=2 VALID conv + relu (encoder; tiny fraction of total work)
// ---------------------------------------------------------------------------
__global__ void conv_k4s2_relu(const float* __restrict__ x, const float* __restrict__ w,
                               const float* __restrict__ b, float* __restrict__ y,
                               int Ci, int Hi, int Wi, int Co, int Ho, int Wo)
{
  int idx = blockIdx.x * blockDim.x + threadIdx.x;
  int total = NN * Co * Ho * Wo;
  if (idx >= total) return;
  int wo = idx % Wo; int t = idx / Wo;
  int ho = t % Ho;   t /= Ho;
  int co = t % Co;   int n = t / Co;
  float acc = b[co];
  const float* xb = x + (size_t)n * Ci * Hi * Wi + ho * 2 * Wi + wo * 2;
  const float* wb = w + (size_t)co * Ci * 16;
  for (int ci = 0; ci < Ci; ++ci) {
    const float* xc = xb + (size_t)ci * Hi * Wi;
    const float* wc = wb + ci * 16;
#pragma unroll
    for (int ky = 0; ky < 4; ++ky)
#pragma unroll
      for (int kx = 0; kx < 4; ++kx)
        acc = fmaf(xc[ky * Wi + kx], wc[ky * 4 + kx], acc);
  }
  y[idx] = fmaxf(acc, 0.f);
}

// [64,256,2,2] -> avg-pool -> f16 A matrix [64,256]
__global__ void avgpool_pack(const float* __restrict__ x, h16* __restrict__ a)
{
  int idx = blockIdx.x * blockDim.x + threadIdx.x;
  if (idx >= NN * 256) return;
  const float* p = x + (size_t)idx * 4;
  a[idx] = (h16)(0.25f * (p[0] + p[1] + p[2] + p[3]));
}

// ---------------------------------------------------------------------------
// RSSM helpers
// ---------------------------------------------------------------------------
__global__ void init_state(float* __restrict__ stoch, float* __restrict__ deter,
                           h16* __restrict__ postin_pad)
{
  int idx = blockIdx.x * blockDim.x + threadIdx.x;
  if (idx < BB * LL * KK) stoch[idx] = ((idx & (KK - 1)) == 0) ? 1.f : 0.f;
  if (idx < BB * DD) deter[idx] = 0.f;
  if (idx < 8 * FEAT) postin_pad[idx] = (h16)0.f;   // rows 8..15 of postin
}

// xt f16 [16,1056] = [stoch | one-hot(action) | pad]
__global__ void build_xt(h16* __restrict__ xt, const float* __restrict__ stoch,
                         const int* __restrict__ actions, int t)
{
  int idx = blockIdx.x * blockDim.x + threadIdx.x;
  if (idx >= 16 * 1056) return;
  int row = idx / 1056, col = idx - row * 1056;
  float v = 0.f;
  if (row < BB) {
    if (col < LL * KK)            v = stoch[row * LL * KK + col];
    else if (col < LL * KK + AA)  v = (actions[t * BB + row] == col - LL * KK) ? 1.f : 0.f;
  }
  xt[idx] = (h16)v;
}

// GRU combine + build posterior input + write deter half of features
__global__ void gru_combine(const float* __restrict__ gi, const float* __restrict__ gh,
                            float* __restrict__ deter, h16* __restrict__ hf16,
                            h16* __restrict__ postin, const float* __restrict__ embed,
                            float* __restrict__ feat_out, int t)
{
  int idx = blockIdx.x * blockDim.x + threadIdx.x;
  if (idx >= BB * FEAT) return;
  int b = idx / FEAT, col = idx - b * FEAT;
  if (col < DD) {
    int d = col;
    float r = 1.f / (1.f + expf(-(gi[b * 1536 + d] + gh[b * 1536 + d])));
    float z = 1.f / (1.f + expf(-(gi[b * 1536 + 512 + d] + gh[b * 1536 + 512 + d])));
    float nn = tanhf(gi[b * 1536 + 1024 + d] + r * gh[b * 1536 + 1024 + d]);
    float h = (1.f - z) * nn + z * deter[b * DD + d];
    deter[b * DD + d] = h;
    hf16[b * DD + d] = (h16)h;
    postin[b * FEAT + d] = (h16)h;
    feat_out[(size_t)(t * BB + b) * FEAT + d] = h;
  } else {
    postin[b * FEAT + col] = (h16)embed[(size_t)(t * BB + b) * EE + (col - DD)];
  }
}

// stoch = one-hot(argmax(post_logits + g)); also writes stoch half of features
__global__ void gumbel_hard(const float* __restrict__ post_logits,
                            const float* __restrict__ gum,
                            float* __restrict__ stoch, float* __restrict__ feat_out, int t)
{
  int idx = blockIdx.x * blockDim.x + threadIdx.x;   // one thread per (b,l)
  if (idx >= BB * LL) return;
  int b = idx / LL, l = idx - b * LL;
  const float* lg = post_logits + ((size_t)(t * BB + b) * LL + l) * KK;
  const float* g  = gum         + ((size_t)(t * BB + b) * LL + l) * KK;
  int am = 0; float best = lg[0] + g[0];
  for (int k = 1; k < KK; ++k) {
    float v = lg[k] + g[k];
    if (v > best) { best = v; am = k; }
  }
  for (int k = 0; k < KK; ++k) {
    float v = (k == am) ? 1.f : 0.f;
    stoch[b * LL * KK + l * KK + k] = v;
    feat_out[(size_t)(t * BB + b) * FEAT + DD + l * KK + k] = v;
  }
}

// ---------------------------------------------------------------------------
// Decoder repack kernels (deconv k=s=4 => pixel-blocked GEMMs, no overlap)
// ---------------------------------------------------------------------------
// y1 [64,2048]=[n][oc*16+p] -> A2 [1024,128]: A2[n*16+p][ic] = y1[n][ic*16+p]
__global__ void pack_a2(const float* __restrict__ y1, h16* __restrict__ a2)
{
  int idx = blockIdx.x * blockDim.x + threadIdx.x;
  if (idx >= 1024 * 128) return;
  int m = idx >> 7, ic = idx & 127;
  int n = m >> 4, p = m & 15;
  a2[idx] = (h16)y1[(size_t)n * 2048 + ic * 16 + p];
}
// g2 [1024,1024] rows (n,iy,ix), cols oc*16+ky*4+kx -> A3 [16384,64] rows (n,oy,ox)
__global__ void pack_a3(const float* __restrict__ g2, h16* __restrict__ a3)
{
  int idx = blockIdx.x * blockDim.x + threadIdx.x;
  if (idx >= 16384 * 64) return;
  int m = idx >> 6, oc = idx & 63;
  int n = m >> 8, rem = m & 255, oy = rem >> 4, ox = rem & 15;
  int row = n * 16 + (oy >> 2) * 4 + (ox >> 2);
  int col = oc * 16 + (oy & 3) * 4 + (ox & 3);
  a3[idx] = (h16)g2[(size_t)row * 1024 + col];
}
// g3 [16384,512] rows (n,iy,ix in 16x16), cols oc*16+ky*4+kx -> A4 [262144,32]
__global__ void pack_a4(const float* __restrict__ g3, h16* __restrict__ a4)
{
  int idx = blockIdx.x * blockDim.x + threadIdx.x;
  if (idx >= 262144 * 32) return;
  int m = idx >> 5, oc = idx & 31;
  int n = m >> 12, rem = m & 4095, y = rem >> 6, x = rem & 63;
  int row = n * 256 + (y >> 2) * 16 + (x >> 2);
  int col = oc * 16 + (y & 3) * 4 + (x & 3);
  a4[idx] = (h16)g3[(size_t)row * 512 + col];
}

// continue head: sigmoid(feat . c_w + c_b), 64 rows
__global__ void cont_head(const float* __restrict__ feat, const float* __restrict__ cw,
                          const float* __restrict__ cb, float* __restrict__ out)
{
  int n = blockIdx.x * blockDim.x + threadIdx.x;
  if (n >= NN) return;
  float acc = cb[0];
  const float* f = feat + (size_t)n * FEAT;
  for (int k = 0; k < FEAT; ++k) acc = fmaf(f[k], cw[k], acc);
  out[n] = 1.f / (1.f + expf(-acc));
}

// obs_target = int(obs*255), stored as float
__global__ void obs_target(const float* __restrict__ obs, float* __restrict__ out)
{
  int idx = blockIdx.x * blockDim.x + threadIdx.x;
  if (idx >= NN * CC * HH * WW) return;
  out[idx] = (float)(int)(obs[idx] * 255.f);
}

// ---------------------------------------------------------------------------
// Host orchestration
// ---------------------------------------------------------------------------
extern "C" void kernel_launch(void* const* d_in, const int* in_sizes, int n_in,
                              void* d_out, int out_size, void* d_ws, size_t ws_size,
                              hipStream_t stream)
{
  (void)in_sizes; (void)n_in; (void)out_size; (void)ws_size;
  // ---- inputs (setup_inputs dict order, params flattened in insertion order)
  const float* obs   = (const float*)d_in[0];
  const int*   acts  = (const int*)  d_in[1];
  const float* gum   = (const float*)d_in[2];
  const float* e_c1 = (const float*)d_in[3],  *e_b1 = (const float*)d_in[4];
  const float* e_c2 = (const float*)d_in[5],  *e_b2 = (const float*)d_in[6];
  const float* e_c3 = (const float*)d_in[7],  *e_b3 = (const float*)d_in[8];
  const float* e_c4 = (const float*)d_in[9],  *e_b4 = (const float*)d_in[10];
  const float* e_fcw = (const float*)d_in[11], *e_fcb = (const float*)d_in[12];
  const float* gru_wih = (const float*)d_in[13], *gru_bih = (const float*)d_in[14];
  const float* gru_whh = (const float*)d_in[15], *gru_bhh = (const float*)d_in[16];
  const float* pr_w1 = (const float*)d_in[17], *pr_b1 = (const float*)d_in[18];
  const float* pr_w2 = (const float*)d_in[19], *pr_b2 = (const float*)d_in[20];
  const float* po_w1 = (const float*)d_in[21], *po_b1 = (const float*)d_in[22];
  const float* po_w2 = (const float*)d_in[23], *po_b2 = (const float*)d_in[24];
  const float* d_fcw = (const float*)d_in[25], *d_fcb = (const float*)d_in[26];
  const float* d_t1 = (const float*)d_in[27], *d_tb1 = (const float*)d_in[28];
  const float* d_t2 = (const float*)d_in[29], *d_tb2 = (const float*)d_in[30];
  const float* d_t3 = (const float*)d_in[31], *d_tb3 = (const float*)d_in[32];
  const float* d_c  = (const float*)d_in[33], *d_cb  = (const float*)d_in[34];
  const float* r_w  = (const float*)d_in[35], *r_b   = (const float*)d_in[36];
  const float* c_w  = (const float*)d_in[37], *c_b   = (const float*)d_in[38];

  float* out = (float*)d_out;
  // ---- output offsets (floats), return order of reference()
  const size_t po_prior  = 0;
  const size_t po_post   = po_prior + (size_t)TT * BB * LL * KK;      // 65536
  const size_t po_feat   = po_post  + (size_t)TT * BB * LL * KK;      // 131072
  const size_t po_recon  = po_feat  + (size_t)TT * BB * FEAT;         // 229376
  const size_t po_reward = po_recon + (size_t)NN * CC * HH * WW * BINS;
  const size_t po_cont   = po_reward + (size_t)NN * BINS;
  const size_t po_obst   = po_cont + NN;

  // ---- workspace bump allocator
  char* base = (char*)d_ws;
  size_t off = 0;
  auto alloc = [&](size_t bytes) -> void* {
    off = (off + 255) & ~(size_t)255;
    void* p = base + off; off += bytes; return p;
  };
  // f16 weights (padded [Np,Kp])
  h16* wih16  = (h16*)alloc(1536 * 1056 * 2);
  h16* whh16  = (h16*)alloc(1536 * 512 * 2);
  h16* efcw16 = (h16*)alloc(1024 * 256 * 2);
  h16* prw116 = (h16*)alloc(208 * 512 * 2);
  h16* prw216 = (h16*)alloc(1024 * 224 * 2);
  h16* pow116 = (h16*)alloc(208 * 1536 * 2);
  h16* pow216 = (h16*)alloc(1024 * 224 * 2);
  h16* dfcw16 = (h16*)alloc(256 * 1536 * 2);
  h16* w1t16  = (h16*)alloc(2048 * 256 * 2);
  h16* w2t16  = (h16*)alloc(1024 * 128 * 2);
  h16* w3t16  = (h16*)alloc(512 * 64 * 2);
  h16* wc16   = (h16*)alloc(768 * 32 * 2);
  h16* rw16   = (h16*)alloc(256 * 1536 * 2);
  // activations
  float* x1   = (float*)alloc((size_t)NN * 32 * 31 * 31 * 4);
  float* x2   = (float*)alloc((size_t)NN * 64 * 14 * 14 * 4);
  float* x3   = (float*)alloc((size_t)NN * 128 * 6 * 6 * 4);
  float* x4   = (float*)alloc((size_t)NN * 256 * 2 * 2 * 4);
  h16*   apool = (h16*)alloc(NN * 256 * 2);
  float* embed = (float*)alloc((size_t)NN * EE * 4);
  float* stoch = (float*)alloc(BB * LL * KK * 4);
  float* deter = (float*)alloc(BB * DD * 4);
  h16*   xt    = (h16*)alloc(16 * 1056 * 2);
  h16*   hf16  = (h16*)alloc(16 * 512 * 2);
  h16*   postin = (h16*)alloc(16 * 1536 * 2);
  float* gi    = (float*)alloc(16 * 1536 * 4);
  float* gh    = (float*)alloc(16 * 1536 * 4);
  float* ph    = (float*)alloc(16 * 200 * 4);
  h16*   phh   = (h16*)alloc(16 * 224 * 2);
  h16*   feat16 = (h16*)alloc((size_t)NN * FEAT * 2);
  float* dfc   = (float*)alloc((size_t)NN * 256 * 4);
  h16*   dfc16 = (h16*)alloc((size_t)NN * 256 * 2);
  float* y1    = (float*)alloc((size_t)NN * 2048 * 4);
  h16*   a2    = (h16*)alloc((size_t)1024 * 128 * 2);
  float* g2    = (float*)alloc((size_t)1024 * 1024 * 4);
  h16*   a3    = (h16*)alloc((size_t)16384 * 64 * 2);
  float* g3    = (float*)alloc((size_t)16384 * 512 * 4);
  h16*   a4    = (h16*)alloc((size_t)262144 * 32 * 2);

  auto ew = [&](int total) { return dim3(CEILDIV(total, 256)); };
  auto gemm = [&](const h16* A, int lda, const h16* Wm, int ldw,
                  const float* bias, int bshift, float* Out, int ldo,
                  int M, int N, int Kp, int act) {
    int waves = CEILDIV(M, 16) * CEILDIV(N, 16);
    hipLaunchKernelGGL(gemm_wmma, dim3(CEILDIV(waves, 8)), dim3(256), 0, stream,
                       A, lda, Wm, ldw, bias, bshift, Out, ldo, M, N, Kp, act);
  };

  // ---- pack all weights to f16 (once per call; small)
  struct PW { const float* s; h16* d; int Nn, Kd, Np, Kp, sn, sk; };
  const PW pws[] = {
    { gru_wih, wih16, 1536, 1042, 1536, 1056, 1042, 1 },
    { gru_whh, whh16, 1536, 512, 1536, 512, 512, 1 },
    { e_fcw, efcw16, 1024, 256, 1024, 256, 256, 1 },
    { pr_w1, prw116, 200, 512, 208, 512, 512, 1 },
    { pr_w2, prw216, 1024, 200, 1024, 224, 200, 1 },
    { po_w1, pow116, 200, 1536, 208, 1536, 1536, 1 },
    { po_w2, pow216, 1024, 200, 1024, 224, 200, 1 },
    { d_fcw, dfcw16, 256, 1536, 256, 1536, 1536, 1 },
    { d_t1, w1t16, 2048, 256, 2048, 256, 1, 2048 },   // (ic,oc,4,4): n=oc*16+p, k=ic
    { d_t2, w2t16, 1024, 128, 1024, 128, 1, 1024 },
    { d_t3, w3t16, 512, 64, 512, 64, 1, 512 },
    { d_c, wc16, 765, 32, 768, 32, 32, 1 },
    { r_w, rw16, 255, 1536, 256, 1536, 1536, 1 },
  };
  for (const PW& p : pws)
    hipLaunchKernelGGL(pack_w, ew(p.Np * p.Kp), dim3(256), 0, stream,
                       p.s, p.d, p.Nn, p.Kd, p.Np, p.Kp, p.sn, p.sk);

  // ---- encoder ----
  hipLaunchKernelGGL(conv_k4s2_relu, ew(NN * 32 * 31 * 31), dim3(256), 0, stream,
                     obs, e_c1, e_b1, x1, CC, 64, 64, 32, 31, 31);
  hipLaunchKernelGGL(conv_k4s2_relu, ew(NN * 64 * 14 * 14), dim3(256), 0, stream,
                     x1, e_c2, e_b2, x2, 32, 31, 31, 64, 14, 14);
  hipLaunchKernelGGL(conv_k4s2_relu, ew(NN * 128 * 6 * 6), dim3(256), 0, stream,
                     x2, e_c3, e_b3, x3, 64, 14, 14, 128, 6, 6);
  hipLaunchKernelGGL(conv_k4s2_relu, ew(NN * 256 * 2 * 2), dim3(256), 0, stream,
                     x3, e_c4, e_b4, x4, 128, 6, 6, 256, 2, 2);
  hipLaunchKernelGGL(avgpool_pack, ew(NN * 256), dim3(256), 0, stream, x4, apool);
  gemm(apool, 256, efcw16, 256, e_fcb, 0, embed, EE, NN, EE, 256, 0);

  // ---- RSSM scan ----
  hipLaunchKernelGGL(init_state, ew(8 * FEAT), dim3(256), 0, stream,
                     stoch, deter, postin + 8 * FEAT);
  for (int t = 0; t < TT; ++t) {
    hipLaunchKernelGGL(build_xt, ew(16 * 1056), dim3(256), 0, stream, xt, stoch, acts, t);
    gemm(xt, 1056, wih16, 1056, gru_bih, 0, gi, 1536, BB, 1536, 1056, 0);
    hipLaunchKernelGGL(cvt_pad, ew(16 * 512), dim3(256), 0, stream,
                       deter, BB, 512, 512, hf16, 16, 512);
    gemm(hf16, 512, whh16, 512, gru_bhh, 0, gh, 1536, BB, 1536, 512, 0);
    hipLaunchKernelGGL(gru_combine, ew(BB * FEAT), dim3(256), 0, stream,
                       gi, gh, deter, hf16, postin, embed, out + po_feat, t);
    // prior
    gemm(hf16, 512, prw116, 512, pr_b1, 0, ph, 200, BB, 200, 512, 1);
    hipLaunchKernelGGL(cvt_pad, ew(16 * 224), dim3(256), 0, stream,
                       ph, BB, 200, 200, phh, 16, 224);
    gemm(phh, 224, prw216, 224, pr_b2, 0, out + po_prior + (size_t)t * BB * LL * KK,
         LL * KK, BB, LL * KK, 224, 0);
    // posterior
    gemm(postin, 1536, pow116, 1536, po_b1, 0, ph, 200, BB, 200, 1536, 1);
    hipLaunchKernelGGL(cvt_pad, ew(16 * 224), dim3(256), 0, stream,
                       ph, BB, 200, 200, phh, 16, 224);
    gemm(phh, 224, pow216, 224, po_b2, 0, out + po_post + (size_t)t * BB * LL * KK,
         LL * KK, BB, LL * KK, 224, 0);
    hipLaunchKernelGGL(gumbel_hard, ew(BB * LL), dim3(256), 0, stream,
                       out + po_post, gum, stoch, out + po_feat, t);
  }

  // ---- decoder ----
  hipLaunchKernelGGL(cvt_pad, ew(NN * FEAT), dim3(256), 0, stream,
                     out + po_feat, NN, FEAT, FEAT, feat16, NN, FEAT);
  gemm(feat16, FEAT, dfcw16, FEAT, d_fcb, 0, dfc, 256, NN, 256, FEAT, 0);
  hipLaunchKernelGGL(cvt_pad, ew(NN * 256), dim3(256), 0, stream,
                     dfc, NN, 256, 256, dfc16, NN, 256);
  gemm(dfc16, 256, w1t16, 256, d_tb1, 4, y1, 2048, NN, 2048, 256, 1);       // deconv1
  hipLaunchKernelGGL(pack_a2, ew(1024 * 128), dim3(256), 0, stream, y1, a2);
  gemm(a2, 128, w2t16, 128, d_tb2, 4, g2, 1024, 1024, 1024, 128, 1);        // deconv2
  hipLaunchKernelGGL(pack_a3, ew(16384 * 64), dim3(256), 0, stream, g2, a3);
  gemm(a3, 64, w3t16, 64, d_tb3, 4, g3, 512, 16384, 512, 64, 1);            // deconv3
  hipLaunchKernelGGL(pack_a4, ew(262144 * 32), dim3(256), 0, stream, g3, a4);
  {
    int waves = (262144 / 16) * (48 / RNT);
    hipLaunchKernelGGL(gemm_recon, dim3(CEILDIV(waves, 8)), dim3(256), 0, stream,
                       a4, wc16, d_cb, out + po_recon);
  }

  // ---- heads ----
  gemm(feat16, FEAT, rw16, FEAT, r_b, 0, out + po_reward, BINS, NN, BINS, FEAT, 0);
  hipLaunchKernelGGL(cont_head, ew(NN), dim3(256), 0, stream,
                     out + po_feat, c_w, c_b, out + po_cont);
  hipLaunchKernelGGL(obs_target, ew(NN * CC * HH * WW), dim3(256), 0, stream,
                     obs, out + po_obst);
}
